// AutoformerEncoderLayer_90469191123604
// MI455X (gfx1250) — matrix operand order repaired and checked by reading, not compiled, not run on hardware
//
#include <hip/hip_runtime.h>
#include <hip/hip_bf16.h>
#include <hip/hip_fp16.h>

// ---------------------------------------------------------------------------
// Autoformer encoder layer for MI455X (gfx1250, wave32, WMMA).
//  * FFT auto-correlation computed exactly as circular-diagonal sums of Q.K^T
//    -> v_wmma_f32_16x16x32_f16 hot loops.
//  * K tiles staged to LDS with GLOBAL_LOAD_ASYNC_TO_LDS_B128 (ASYNCcnt path).
//  * Weights pre-transposed to [N,K] f16 so every WMMA fragment load is a
//    contiguous 16/32-byte vector load.
// ---------------------------------------------------------------------------

typedef __attribute__((ext_vector_type(16))) _Float16 v16h;
typedef __attribute__((ext_vector_type(8)))  _Float16 v8h;
typedef __attribute__((ext_vector_type(8)))  float    v8f;

namespace cfg {
constexpr int B = 4, L = 4096, D = 512, H = 8, DK = 64, F = 2048;
constexpr int M = B * L;            // 16384 rows
constexpr int TOPK = 8;             // int(log(4097)) = 8
constexpr int MAPAD = 12;           // (25-1)/2
}

// --------------------- async load-to-LDS support (gfx1250) ------------------
// Probe round confirmed: builtin exists, 4 args, param0 = AS(1) v4i*
// (clang prints AS1 as '__device__' in HIP).  Param1 is the AS(3) twin.

typedef int v4i __attribute__((vector_size(4 * sizeof(int))));
typedef __attribute__((address_space(1))) v4i as1_v4i;
typedef __attribute__((address_space(3))) v4i as3_v4i;

#if defined(__has_builtin)
#if __has_builtin(__builtin_amdgcn_global_load_async_to_lds_b128)
#define USE_ASYNC_LDS 1
#endif
#endif
#ifndef USE_ASYNC_LDS
#define USE_ASYNC_LDS 0
#endif

__device__ __forceinline__ void wait_asynccnt0() {
#if defined(__has_builtin)
#if __has_builtin(__builtin_amdgcn_s_wait_asynccnt)
  __builtin_amdgcn_s_wait_asynccnt(0);
  return;
#else
  asm volatile("s_wait_asynccnt 0" ::: "memory");
  return;
#endif
#else
  asm volatile("s_wait_asynccnt 0" ::: "memory");
#endif
}

// ----------------------------- elementwise helpers -------------------------

__global__ void f32_to_f16_kernel(const float* __restrict__ in,
                                  _Float16* __restrict__ out, int n) {
  int i = blockIdx.x * blockDim.x + threadIdx.x;
  if (i < n) out[i] = (_Float16)in[i];
}

__global__ void fill_zero_kernel(float* __restrict__ p, int n) {
  int i = blockIdx.x * blockDim.x + threadIdx.x;
  if (i < n) p[i] = 0.0f;
}

// Transpose+convert: in [K,N] f32 row-major -> out [N,K] f16 row-major.
__global__ __launch_bounds__(256)
void transpose_f32_to_f16_kernel(const float* __restrict__ in,
                                 _Float16* __restrict__ out, int K, int N) {
  __shared__ float tile[32][33];
  const int kt = blockIdx.x * 32, nt = blockIdx.y * 32;
  const int tx = threadIdx.x & 31, ty = threadIdx.x >> 5;   // 32 x 8
  for (int r = ty; r < 32; r += 8)
    tile[r][tx] = in[(size_t)(kt + r) * N + (nt + tx)];
  __syncthreads();
  for (int r = ty; r < 32; r += 8)
    out[(size_t)(nt + r) * K + (kt + tx)] = (_Float16)tile[tx][r];
}

// ----------------------------- WMMA GEMM -----------------------------------
// C[m,n] = sum_k A16[m,k] * Bt[n,k] + bias[n]   (Bt = B transposed, [N,K])
// Block = 256 threads = 8 waves arranged 2(M) x 4(N); each wave computes a
// 16x64 strip = 4 accumulators, so the A fragment is reused 4x and every
// fragment load is a contiguous 16B/32B vector load (k = 16*half + s order in A and B).
// EPI 0: f32 row-major.  EPI 1: f16 scatter to [B,H,L,dk].  EPI 2: GELU->f16.

template <int EPI>
__global__ __launch_bounds__(256)
void gemm_wmma_kernel(const _Float16* __restrict__ A,
                      const _Float16* __restrict__ Bt,
                      const float* __restrict__ bias,
                      void* __restrict__ out, int Mdim, int N, int K) {
  const int wave = threadIdx.x >> 5;
  const int lane = threadIdx.x & 31;
  const int l15  = lane & 15;
  const int half = lane >> 4;
  const int m0 = blockIdx.x * 32 + (wave >> 2) * 16;
  const int n0 = blockIdx.y * 256 + (wave & 3) * 64;
  if (m0 >= Mdim || n0 >= N) return;          // uniform per wave

  v8f c0 = {}, c1 = {}, c2 = {}, c3 = {};
  const _Float16* arow = A + (size_t)(m0 + l15) * K;
  const int koff = half * 16;                 // A: lanes16-31 hold K+16
  const _Float16* b0p = Bt + (size_t)(n0 + l15) * K + half * 16;
  const _Float16* b1p = b0p + (size_t)16 * K;
  const _Float16* b2p = b0p + (size_t)32 * K;
  const _Float16* b3p = b0p + (size_t)48 * K;

  for (int kb = 0; kb < K; kb += 32) {
    v8h alo = *(const v8h*)(arow + kb + koff);
    v8h ahi = *(const v8h*)(arow + kb + 8 + koff);
    v16h a;
#pragma unroll
    for (int i = 0; i < 8; ++i) { a[i] = alo[i]; a[8 + i] = ahi[i]; }
    v16h b0 = *(const v16h*)(b0p + kb);
    v16h b1 = *(const v16h*)(b1p + kb);
    v16h b2 = *(const v16h*)(b2p + kb);
    v16h b3 = *(const v16h*)(b3p + kb);
    c0 = __builtin_amdgcn_wmma_f32_16x16x32_f16(false, a, false, b0, (short)0, c0, false, false);
    c1 = __builtin_amdgcn_wmma_f32_16x16x32_f16(false, a, false, b1, (short)0, c1, false, false);
    c2 = __builtin_amdgcn_wmma_f32_16x16x32_f16(false, a, false, b2, (short)0, c2, false, false);
    c3 = __builtin_amdgcn_wmma_f32_16x16x32_f16(false, a, false, b3, (short)0, c3, false, false);
  }

#pragma unroll
  for (int j = 0; j < 4; ++j) {
    const v8f& c = (j == 0) ? c0 : (j == 1) ? c1 : (j == 2) ? c2 : c3;
#pragma unroll
    for (int r = 0; r < 8; ++r) {
      const int m = m0 + r + half * 8;        // C/D layout: vgpr r, lane-half
      const int n = n0 + j * 16 + l15;
      float v = c[r] + bias[n];
      if (EPI == 0) {
        ((float*)out)[(size_t)m * N + n] = v;
      } else if (EPI == 1) {
        const int b_ = m >> 12, l_ = m & (cfg::L - 1);
        const int h_ = n >> 6,  d_ = n & (cfg::DK - 1);
        ((_Float16*)out)[(((size_t)(b_ * cfg::H + h_) * cfg::L + l_) * cfg::DK) + d_] =
            (_Float16)v;
      } else {
        float g = 0.5f * v * (1.0f + erff(v * 0.70710678118654752f));
        ((_Float16*)out)[(size_t)m * N + n] = (_Float16)g;
      }
    }
  }
}

// ----------------------------- auto-correlation ----------------------------
// corr[b,h,tau] = (1/dk) * sum over the tau-th circular diagonal of S=Q.K^T.
// One block per (bh, 16-row i-tile).  K rows streamed through LDS in 128-row
// chunks (async load-to-LDS); 8 waves each compute 2 j-tiles per chunk;
// diagonals accumulate in an LDS histogram, flushed with global atomics.

__global__ __launch_bounds__(256)
void corr_wmma_kernel(const _Float16* __restrict__ Q,
                      const _Float16* __restrict__ Kmat,
                      float* __restrict__ corr) {
  using namespace cfg;
  constexpr int CH = 128;                     // K rows per LDS chunk
  constexpr int KP = DK + 8;                  // padded LDS row (halves)
  __shared__ float    lcorr[L];               // 16 KB
  __shared__ _Float16 Ks[CH * KP];            // 18 KB

  const int bh = blockIdx.y;
  const int i0 = blockIdx.x * 16;
  const _Float16* Qb = Q    + (size_t)bh * L * DK;
  const _Float16* Kb = Kmat + (size_t)bh * L * DK;

  for (int t = threadIdx.x; t < L; t += 256) lcorr[t] = 0.0f;

  const int wave = threadIdx.x >> 5;
  const int lane = threadIdx.x & 31;
  const int l15  = lane & 15;
  const int half = lane >> 4;
  const int koff = half * 8;

  // Preload both K-steps (dk = 64 = 2 x 32) of the A (Q) fragment.
  v16h a0, a1;
  {
    const _Float16* arow = Qb + (size_t)(i0 + l15) * DK;
    v8h x0 = *(const v8h*)(arow + 0  + koff);
    v8h x1 = *(const v8h*)(arow + 16 + koff);
    v8h x2 = *(const v8h*)(arow + 32 + koff);
    v8h x3 = *(const v8h*)(arow + 48 + koff);
#pragma unroll
    for (int i = 0; i < 8; ++i) {
      a0[i] = x0[i]; a0[8 + i] = x1[i];
      a1[i] = x2[i]; a1[8 + i] = x3[i];
    }
  }

  // Staging split: thread t owns 32 halves (64B) = half of one K row.
  const int srow = threadIdx.x >> 1;
  const int scol = (threadIdx.x & 1) * 32;

  for (int c = 0; c < L / CH; ++c) {
    __syncthreads();                          // Ks free, lcorr init done
    const _Float16* src = Kb + ((size_t)c * CH + srow) * DK + scol;
    _Float16*       dst = &Ks[srow * KP + scol];
#if USE_ASYNC_LDS
#pragma unroll
    for (int i = 0; i < 4; ++i)
      __builtin_amdgcn_global_load_async_to_lds_b128(
          (as1_v4i*)(src + i * 8), (as3_v4i*)(dst + i * 8), 0, 0);
    wait_asynccnt0();
#else
#pragma unroll
    for (int i = 0; i < 4; ++i)
      *(v8h*)(dst + i * 8) = *(const v8h*)(src + i * 8);
#endif
    __syncthreads();

#pragma unroll
    for (int jj = 0; jj < 2; ++jj) {
      const int jt = wave * 2 + jj;           // 16 j-tiles per chunk / 8 waves
      const int jl = jt * 16;
      const _Float16* brow = &Ks[(jl + l15) * KP + half * 8];
      v16h b0, b1;
      {
        v8h p0 = *(const v8h*)(brow);
        v8h p1 = *(const v8h*)(brow + 16);
        v8h p2 = *(const v8h*)(brow + 32);
        v8h p3 = *(const v8h*)(brow + 48);
#pragma unroll
        for (int i = 0; i < 8; ++i) {
          b0[i] = p0[i]; b0[8 + i] = p1[i];
          b1[i] = p2[i]; b1[8 + i] = p3[i];
        }
      }
      v8f cacc = {};
      cacc = __builtin_amdgcn_wmma_f32_16x16x32_f16(false, a0, false, b0, (short)0, cacc, false, false);
      cacc = __builtin_amdgcn_wmma_f32_16x16x32_f16(false, a1, false, b1, (short)0, cacc, false, false);
      const int j0g = c * CH + jl;
#pragma unroll
      for (int r = 0; r < 8; ++r) {
        const int ig = i0 + r + half * 8;
        const int jg = j0g + l15;
        atomicAdd(&lcorr[(ig - jg) & (L - 1)], cacc[r]);
      }
    }
  }
  __syncthreads();

  float* cg = corr + (size_t)bh * L;
  for (int t = threadIdx.x; t < L; t += 256)
    atomicAdd(&cg[t], lcorr[t] * (1.0f / DK));
}

// ----------------------------- top-k + softmax -----------------------------

__global__ __launch_bounds__(256)
void topk_softmax_kernel(const float* __restrict__ corr,
                         float* __restrict__ tw, int* __restrict__ ti) {
  using namespace cfg;
  __shared__ float sv[256];
  __shared__ int   si[256];
  __shared__ int   chosen[TOPK];
  __shared__ float cval[TOPK];
  const int bh = blockIdx.x;
  const float* c = corr + (size_t)bh * L;

  for (int k = 0; k < TOPK; ++k) {
    float best = -3.4e38f; int bi = -1;
    for (int t = threadIdx.x; t < L; t += 256) {
      bool skip = false;
      for (int p = 0; p < k; ++p) skip |= (chosen[p] == t);
      float v = c[t];
      if (!skip && v > best) { best = v; bi = t; }
    }
    sv[threadIdx.x] = best; si[threadIdx.x] = bi;
    __syncthreads();
    for (int s = 128; s > 0; s >>= 1) {
      if (threadIdx.x < s && sv[threadIdx.x + s] > sv[threadIdx.x]) {
        sv[threadIdx.x] = sv[threadIdx.x + s];
        si[threadIdx.x] = si[threadIdx.x + s];
      }
      __syncthreads();
    }
    if (threadIdx.x == 0) { chosen[k] = si[0]; cval[k] = sv[0]; }
    __syncthreads();
  }
  if (threadIdx.x == 0) {
    float mx = cval[0];
    for (int k = 1; k < TOPK; ++k) mx = fmaxf(mx, cval[k]);
    float e[TOPK], s = 0.0f;
    for (int k = 0; k < TOPK; ++k) { e[k] = __expf(cval[k] - mx); s += e[k]; }
    for (int k = 0; k < TOPK; ++k) {
      tw[bh * TOPK + k] = e[k] / s;
      ti[bh * TOPK + k] = chosen[k];
    }
  }
}

// ----------------------------- roll-gather of V ----------------------------

__global__ __launch_bounds__(256)
void gather_attn_kernel(const _Float16* __restrict__ V,
                        const float* __restrict__ tw,
                        const int* __restrict__ ti,
                        _Float16* __restrict__ outp) {
  using namespace cfg;
  const size_t idx = (size_t)blockIdx.x * 256 + threadIdx.x;
  if (idx >= (size_t)M * D) return;
  const int n = (int)(idx & (D - 1));
  const size_t m = idx >> 9;
  const int l = (int)(m & (L - 1));
  const int b = (int)(m >> 12);
  const int h = n >> 6, d = n & (DK - 1);
  const int bh = b * H + h;
  const _Float16* Vb = V + (size_t)bh * L * DK;
  float acc = 0.0f;
#pragma unroll
  for (int k = 0; k < TOPK; ++k) {
    const int tau = ti[bh * TOPK + k];
    const float w = tw[bh * TOPK + k];
    acc += w * (float)Vb[(size_t)((l + tau) & (L - 1)) * DK + d];
  }
  outp[idx] = (_Float16)acc;
}

// ----------------------------- add + layernorm -----------------------------

__global__ __launch_bounds__(256)
void add_layernorm_kernel(const float* __restrict__ xa,
                          const float* __restrict__ xb,
                          const float* __restrict__ g,
                          const float* __restrict__ be,
                          float* __restrict__ out) {
  using namespace cfg;
  __shared__ float red[256];
  __shared__ float s_mu, s_rstd;
  const int row = blockIdx.x, t = threadIdx.x;
  const float* pa = xa + (size_t)row * D;
  const float* pb = xb + (size_t)row * D;
  const float v0 = pa[t] + pb[t];
  const float v1 = pa[t + 256] + pb[t + 256];

  red[t] = v0 + v1; __syncthreads();
  for (int s = 128; s > 0; s >>= 1) { if (t < s) red[t] += red[t + s]; __syncthreads(); }
  if (t == 0) s_mu = red[0] * (1.0f / D);
  __syncthreads();
  const float mu = s_mu;
  const float d0 = v0 - mu, d1 = v1 - mu;
  red[t] = d0 * d0 + d1 * d1; __syncthreads();
  for (int s = 128; s > 0; s >>= 1) { if (t < s) red[t] += red[t + s]; __syncthreads(); }
  if (t == 0) s_rstd = rsqrtf(red[0] * (1.0f / D) + 1e-5f);
  __syncthreads();
  const float rs = s_rstd;
  out[(size_t)row * D + t]       = d0 * rs * g[t] + be[t];
  out[(size_t)row * D + t + 256] = d1 * rs * g[t + 256] + be[t + 256];
}

// ----------------------------- series decomposition ------------------------

__global__ __launch_bounds__(256)
void decomp_kernel(const float* __restrict__ x, float* __restrict__ seasonal,
                   _Float16* __restrict__ seasonal16, float* __restrict__ trend) {
  using namespace cfg;
  const size_t idx = (size_t)blockIdx.x * 256 + threadIdx.x;
  if (idx >= (size_t)M * D) return;
  const int n = (int)(idx & (D - 1));
  const size_t m = idx >> 9;
  const int l = (int)(m & (L - 1));
  const int b = (int)(m >> 12);
  const float* col = x + (size_t)b * L * D + n;
  float s = 0.0f;
#pragma unroll
  for (int o = -MAPAD; o <= MAPAD; ++o) {
    int ll = l + o;
    ll = ll < 0 ? 0 : (ll > L - 1 ? L - 1 : ll);
    s += col[(size_t)ll * D];
  }
  const float tr = s * (1.0f / 25.0f);
  const float se = x[idx] - tr;
  trend[idx] = tr;
  seasonal[idx] = se;
  seasonal16[idx] = (_Float16)se;
}

__global__ __launch_bounds__(256)
void decomp_final_kernel(const float* __restrict__ x2,
                         const float* __restrict__ trend1,
                         float* __restrict__ seasonal_out,
                         float* __restrict__ trend_out) {
  using namespace cfg;
  const size_t idx = (size_t)blockIdx.x * 256 + threadIdx.x;
  if (idx >= (size_t)M * D) return;
  const int n = (int)(idx & (D - 1));
  const size_t m = idx >> 9;
  const int l = (int)(m & (L - 1));
  const int b = (int)(m >> 12);
  const float* col = x2 + (size_t)b * L * D + n;
  float s = 0.0f;
#pragma unroll
  for (int o = -MAPAD; o <= MAPAD; ++o) {
    int ll = l + o;
    ll = ll < 0 ? 0 : (ll > L - 1 ? L - 1 : ll);
    s += col[(size_t)ll * D];
  }
  const float t2 = s * (1.0f / 25.0f);
  seasonal_out[idx] = x2[idx] - t2;
  trend_out[idx] = trend1[idx] + t2;
}

// ----------------------------- launcher ------------------------------------

extern "C" void kernel_launch(void* const* d_in, const int* in_sizes, int n_in,
                              void* d_out, int out_size, void* d_ws, size_t ws_size,
                              hipStream_t stream) {
  using namespace cfg;
  (void)in_sizes; (void)n_in; (void)out_size; (void)ws_size;

  const float* x   = (const float*)d_in[0];
  const float* Wq  = (const float*)d_in[1];  const float* bq  = (const float*)d_in[2];
  const float* Wk  = (const float*)d_in[3];  const float* bk  = (const float*)d_in[4];
  const float* Wv  = (const float*)d_in[5];  const float* bv  = (const float*)d_in[6];
  const float* Wo  = (const float*)d_in[7];  const float* bo  = (const float*)d_in[8];
  const float* W1  = (const float*)d_in[9];  const float* b1  = (const float*)d_in[10];
  const float* W2  = (const float*)d_in[11]; const float* b2  = (const float*)d_in[12];
  const float* g1  = (const float*)d_in[13]; const float* be1 = (const float*)d_in[14];
  const float* g2  = (const float*)d_in[15]; const float* be2 = (const float*)d_in[16];

  float* out_seasonal = (float*)d_out;
  float* out_trend    = (float*)d_out + (size_t)M * D;

  // --- carve workspace (256-byte aligned slices) ---
  char* ws = (char*)d_ws;
  size_t off = 0;
  auto carve = [&](size_t bytes) -> char* {
    char* p = ws + off;
    off = (off + bytes + 255) & ~(size_t)255;
    return p;
  };
  _Float16* x16   = (_Float16*)carve((size_t)M * D * 2);
  _Float16* wqT   = (_Float16*)carve((size_t)D * D * 2);   // [N,K] f16
  _Float16* wkT   = (_Float16*)carve((size_t)D * D * 2);
  _Float16* wvT   = (_Float16*)carve((size_t)D * D * 2);
  _Float16* woT   = (_Float16*)carve((size_t)D * D * 2);
  _Float16* w1T   = (_Float16*)carve((size_t)D * F * 2);   // [F,D]
  _Float16* w2T   = (_Float16*)carve((size_t)F * D * 2);   // [D,F]
  _Float16* q16   = (_Float16*)carve((size_t)M * D * 2);   // [B,H,L,dk]
  _Float16* k16   = (_Float16*)carve((size_t)M * D * 2);
  _Float16* v16   = (_Float16*)carve((size_t)M * D * 2);
  float*    corr  = (float*)   carve((size_t)B * H * L * 4);
  float*    tw    = (float*)   carve((size_t)B * H * TOPK * 4);
  int*      ti    = (int*)     carve((size_t)B * H * TOPK * 4);
  _Float16* ap16  = (_Float16*)carve((size_t)M * D * 2);   // attn pre-projection
  float*    attn  = (float*)   carve((size_t)M * D * 4);   // reused as ff
  float*    x1    = (float*)   carve((size_t)M * D * 4);   // reused as x2
  float*    tr1   = (float*)   carve((size_t)M * D * 4);
  float*    sea1  = (float*)   carve((size_t)M * D * 4);
  _Float16* sea16 = (_Float16*)carve((size_t)M * D * 2);
  _Float16* h16   = (_Float16*)carve((size_t)M * F * 2);
  float*    ff    = attn;   // attn dead after x1
  float*    x2    = x1;     // x1 dead after decomp1

  const int EW = M * D;
  auto eb = [](int n) { return (n + 255) / 256; };

  // --- staging: x -> f16, weights -> transposed f16 ---
  f32_to_f16_kernel<<<eb(EW), 256, 0, stream>>>(x, x16, EW);
  transpose_f32_to_f16_kernel<<<dim3(D / 32, D / 32), 256, 0, stream>>>(Wq, wqT, D, D);
  transpose_f32_to_f16_kernel<<<dim3(D / 32, D / 32), 256, 0, stream>>>(Wk, wkT, D, D);
  transpose_f32_to_f16_kernel<<<dim3(D / 32, D / 32), 256, 0, stream>>>(Wv, wvT, D, D);
  transpose_f32_to_f16_kernel<<<dim3(D / 32, D / 32), 256, 0, stream>>>(Wo, woT, D, D);
  transpose_f32_to_f16_kernel<<<dim3(D / 32, F / 32), 256, 0, stream>>>(W1, w1T, D, F);
  transpose_f32_to_f16_kernel<<<dim3(F / 32, D / 32), 256, 0, stream>>>(W2, w2T, F, D);

  // --- QKV projections (scatter to [B,H,L,dk] f16) ---
  dim3 gP(M / 32, D / 256);
  gemm_wmma_kernel<1><<<gP, 256, 0, stream>>>(x16, wqT, bq, q16, M, D, D);
  gemm_wmma_kernel<1><<<gP, 256, 0, stream>>>(x16, wkT, bk, k16, M, D, D);
  gemm_wmma_kernel<1><<<gP, 256, 0, stream>>>(x16, wvT, bv, v16, M, D, D);

  // --- auto-correlation via QK^T circular diagonals ---
  fill_zero_kernel<<<eb(B * H * L), 256, 0, stream>>>(corr, B * H * L);
  corr_wmma_kernel<<<dim3(L / 16, B * H), 256, 0, stream>>>(q16, k16, corr);
  topk_softmax_kernel<<<B * H, 256, 0, stream>>>(corr, tw, ti);
  gather_attn_kernel<<<eb(EW), 256, 0, stream>>>(v16, tw, ti, ap16);

  // --- output projection, residual + LN1 ---
  gemm_wmma_kernel<0><<<gP, 256, 0, stream>>>(ap16, woT, bo, attn, M, D, D);
  add_layernorm_kernel<<<M, 256, 0, stream>>>(x, attn, g1, be1, x1);

  // --- decomp 1 ---
  decomp_kernel<<<eb(EW), 256, 0, stream>>>(x1, sea1, sea16, tr1);

  // --- FFN: GELU(s1 @ W1 + b1) @ W2 + b2 ---
  gemm_wmma_kernel<2><<<dim3(M / 32, F / 256), 256, 0, stream>>>(sea16, w1T, b1, h16, M, F, D);
  gemm_wmma_kernel<0><<<gP, 256, 0, stream>>>(h16, w2T, b2, ff, M, D, F);

  // --- residual + LN2, decomp 2, outputs ---
  add_layernorm_kernel<<<M, 256, 0, stream>>>(sea1, ff, g2, be2, x2);
  decomp_final_kernel<<<eb(EW), 256, 0, stream>>>(x2, tr1, out_seasonal, out_trend);
}
